// SegmentSumMLPModule_66554813218851
// MI455X (gfx1250) — matrix operand order repaired and checked by reading, not compiled, not run on hardware
//
#include <hip/hip_runtime.h>
#include <hip/hip_bf16.h>

// ---------------------------------------------------------------------------
// SegmentSumMLP for MI455X (gfx1250): bf16 WMMA (16x16x32) with f32 accum.
// SEG_SIZE==16 == WMMA M-tile, so segment-sum folds into the GEMM epilogue.
// ---------------------------------------------------------------------------

typedef __attribute__((ext_vector_type(16))) __bf16   v16bf;
typedef __attribute__((ext_vector_type(8)))  float    v8f;
typedef __attribute__((ext_vector_type(4)))  unsigned u32x4;

union ABFrag {
    v16bf v;
    u32x4 q[2];
    unsigned short s[16];
};

__device__ inline unsigned short f2bf(float f) {
    unsigned u = __builtin_bit_cast(unsigned, f);
    unsigned r = u + 0x7FFFu + ((u >> 16) & 1u);   // round-to-nearest-even
    return (unsigned short)(r >> 16);
}

__device__ inline v8f vzero8() {
    v8f z = {0.f, 0.f, 0.f, 0.f, 0.f, 0.f, 0.f, 0.f};
    return z;
}

// ---------------------------------------------------------------------------
// Pre-swizzle a [K, N] fp32 weight matrix into bf16 WMMA B-fragments:
//   dst layout: [ktile][ntile][lane][16] bf16, frag = 32 lanes * 32B = 1 KB.
//   B 32x16 bf16 layout: lane n<16 -> col n,    K = kbase + {0..7, 16..23}
//                        lane n>=16 -> col n-16, K = kbase + {8..15, 24..31}
// Rows k >= K are zero (K-padding for IN_DIM=164 -> 192).
// ---------------------------------------------------------------------------
__global__ void swizzleB(const float* __restrict__ W,
                         unsigned short* __restrict__ dst,
                         int K, int Kpad, int N) {
    int idx = blockIdx.x * 256 + threadIdx.x;
    int ntiles = N >> 4;
    int total = (Kpad >> 5) * ntiles * 512;
    if (idx >= total) return;
    int e    = idx & 15;
    int lane = (idx >> 4) & 31;
    int frag = idx >> 9;
    int nt   = frag % ntiles;
    int kt   = frag / ntiles;
    int k    = kt * 32 + ((e >> 3) << 4) + ((lane >> 4) << 3) + (e & 7);
    int col  = nt * 16 + (lane & 15);
    float v  = (k < K) ? W[(long)k * N + col] : 0.f;
    dst[idx] = f2bf(v);
}

// ---------------------------------------------------------------------------
// Encoder: relu(relu(X @ We1 + b1) @ We2 + b2), then per-16-row segment sum.
// Block = 256 threads = 8 waves, handles SEGB=4 segments (4 M-tiles).
// Wave w owns output column tiles {2w, 2w+1} (columns 32w .. 32w+31).
// ---------------------------------------------------------------------------
#define SEGB 4

__global__ __launch_bounds__(256)
void encoder_kernel(const float* __restrict__ feat,
                    const unsigned short* __restrict__ We1s,
                    const unsigned short* __restrict__ We2s,
                    const float* __restrict__ be1,
                    const float* __restrict__ be2,
                    float* __restrict__ sOut) {
    __shared__ __align__(16) unsigned short featL[SEGB][16][192]; // bf16, K padded
    __shared__ __align__(16) unsigned short hL[SEGB][16][256];    // bf16 hidden

    const int  tid     = threadIdx.x;
    const int  lane    = tid & 31;
    const int  w       = tid >> 5;
    const long segBase = (long)blockIdx.x * SEGB;
    const long rowBase = segBase * 16;

    // ---- stage features -> LDS (bf16), coalesced ----
    for (int idx = tid; idx < SEGB * 16 * 164; idx += 256) {
        int r = idx / 164, c = idx % 164;
        featL[r >> 4][r & 15][c] = f2bf(feat[(rowBase + r) * 164 + c]);
    }
    for (int idx = tid; idx < SEGB * 16 * 28; idx += 256) {  // zero K-pad 164..191
        int r = idx / 28, c = idx % 28;
        featL[r >> 4][r & 15][164 + c] = 0;
    }
    __syncthreads();

    const int halfOff = (lane >> 4) << 3;  // A-frag: 0 or 8 within each 16-K half
    const int m       = lane & 15;         // A row / C column-in-tile
    const int mrow    = (lane >> 4) << 3;  // C rows: 0..7 or 8..15

    // ---- GEMM1: [16,192] x [192,256], bias + relu -> hL ----
    for (int ni = 0; ni < 2; ++ni) {
        int nt = w * 2 + ni;
        v8f acc[SEGB];
        for (int s = 0; s < SEGB; ++s) acc[s] = vzero8();
        for (int kt = 0; kt < 6; ++kt) {
            ABFrag b;
            const u32x4* bp = (const u32x4*)(We1s + ((long)(kt * 16 + nt) * 512 + lane * 16));
            b.q[0] = bp[0];
            b.q[1] = bp[1];
            int kb = kt * 32;
            for (int s = 0; s < SEGB; ++s) {
                ABFrag a;
                a.q[0] = *(const u32x4*)&featL[s][m][kb + halfOff];
                a.q[1] = *(const u32x4*)&featL[s][m][kb + 16 + halfOff];
                acc[s] = __builtin_amdgcn_wmma_f32_16x16x32_bf16(
                    false, a.v, false, b.v, (short)0, acc[s], false, false);
            }
        }
        float bias = be1[nt * 16 + m];
        for (int s = 0; s < SEGB; ++s)
            for (int r = 0; r < 8; ++r) {
                float v = acc[s][r] + bias;
                v = v > 0.f ? v : 0.f;
                hL[s][mrow + r][nt * 16 + m] = f2bf(v);
            }
    }
    __syncthreads();

    // ---- GEMM2: [16,256] x [256,256], bias + relu + segment-sum epilogue ----
    for (int ni = 0; ni < 2; ++ni) {
        int nt = w * 2 + ni;
        v8f acc[SEGB];
        for (int s = 0; s < SEGB; ++s) acc[s] = vzero8();
        for (int kt = 0; kt < 8; ++kt) {
            ABFrag b;
            const u32x4* bp = (const u32x4*)(We2s + ((long)(kt * 16 + nt) * 512 + lane * 16));
            b.q[0] = bp[0];
            b.q[1] = bp[1];
            int kb = kt * 32;
            for (int s = 0; s < SEGB; ++s) {
                ABFrag a;
                a.q[0] = *(const u32x4*)&hL[s][m][kb + halfOff];
                a.q[1] = *(const u32x4*)&hL[s][m][kb + 16 + halfOff];
                acc[s] = __builtin_amdgcn_wmma_f32_16x16x32_bf16(
                    false, a.v, false, b.v, (short)0, acc[s], false, false);
            }
        }
        float bias = be2[nt * 16 + m];
        for (int s = 0; s < SEGB; ++s) {
            float colsum = 0.f;
            for (int r = 0; r < 8; ++r) {
                float v = acc[s][r] + bias;
                colsum += (v > 0.f ? v : 0.f);
            }
            colsum += __shfl_xor(colsum, 16, 32);  // combine rows 0-7 and 8-15
            if (lane < 16)
                sOut[(segBase + s) * 256 + nt * 16 + lane] = colsum;
        }
    }
}

// ---------------------------------------------------------------------------
// Head: out = relu(s@Wl0+b)+s ; out = relu(out@Wl1+b)+out ; y = out@Wd + bd.
// Block handles TS=2 tiles of 16 segments (32 segments).
// ---------------------------------------------------------------------------
#define TS 2

__global__ __launch_bounds__(256)
void head_kernel(const float* __restrict__ sIn,
                 const unsigned short* __restrict__ Wl0s,
                 const unsigned short* __restrict__ Wl1s,
                 const float* __restrict__ bl0,
                 const float* __restrict__ bl1,
                 const float* __restrict__ Wd,
                 const float* __restrict__ bd,
                 float* __restrict__ out) {
    __shared__ __align__(16) unsigned short aL[TS][16][256];  // bf16 activation
    __shared__ float sL[TS][16][256];                         // fp32 (residual)

    const int  tid     = threadIdx.x;
    const int  lane    = tid & 31;
    const int  w       = tid >> 5;
    const long rowBase = (long)blockIdx.x * (TS * 16);

    for (int idx = tid; idx < TS * 16 * 256; idx += 256) {
        int r = idx >> 8, c = idx & 255;
        float v = sIn[(rowBase + r) * 256 + c];
        sL[r >> 4][r & 15][c] = v;
        aL[r >> 4][r & 15][c] = f2bf(v);
    }
    __syncthreads();

    const int halfOff = (lane >> 4) << 3;
    const int m       = lane & 15;
    const int mrow    = (lane >> 4) << 3;

    const unsigned short* Wm[2]   = {Wl0s, Wl1s};
    const float*          bias[2] = {bl0, bl1};

    for (int layer = 0; layer < 2; ++layer) {
        v8f acc[2][TS];
        for (int ni = 0; ni < 2; ++ni) {
            int nt = w * 2 + ni;
            for (int t = 0; t < TS; ++t) acc[ni][t] = vzero8();
            for (int kt = 0; kt < 8; ++kt) {
                ABFrag b;
                const u32x4* bp =
                    (const u32x4*)(Wm[layer] + ((long)(kt * 16 + nt) * 512 + lane * 16));
                b.q[0] = bp[0];
                b.q[1] = bp[1];
                int kb = kt * 32;
                for (int t = 0; t < TS; ++t) {
                    ABFrag a;
                    a.q[0] = *(const u32x4*)&aL[t][m][kb + halfOff];
                    a.q[1] = *(const u32x4*)&aL[t][m][kb + 16 + halfOff];
                    acc[ni][t] = __builtin_amdgcn_wmma_f32_16x16x32_bf16(
                        false, a.v, false, b.v, (short)0, acc[ni][t], false, false);
                }
            }
        }
        __syncthreads();  // all aL reads done before rewriting
        for (int ni = 0; ni < 2; ++ni) {
            int   nt = w * 2 + ni;
            float bb = bias[layer][nt * 16 + m];
            for (int t = 0; t < TS; ++t)
                for (int r = 0; r < 8; ++r) {
                    float v = acc[ni][t][r] + bb;
                    v = v > 0.f ? v : 0.f;
                    float o = v + sL[t][mrow + r][nt * 16 + m];  // residual
                    sL[t][mrow + r][nt * 16 + m] = o;
                    aL[t][mrow + r][nt * 16 + m] = f2bf(o);
                }
        }
        __syncthreads();
    }

    // decode: [32 rows] x [256] dot with Wd + bd  (tiny)
    if (tid < TS * 16) {
        int   t = tid >> 4, r = tid & 15;
        float acc = bd[0];
        for (int c = 0; c < 256; ++c) acc += sL[t][r][c] * Wd[c];
        out[rowBase + tid] = acc;
    }
}

// ---------------------------------------------------------------------------
// Workspace layout (bytes):
//   We1s : 0        .. 98304    (6 kt * 16 nt * 1KB)
//   We2s : 98304    .. 229376   (8 * 16 * 1KB)
//   Wl0s : 229376   .. 360448
//   Wl1s : 360448   .. 491520
//   s    : 491520   .. +65536*256*4  (fp32 segment sums)
// ---------------------------------------------------------------------------
extern "C" void kernel_launch(void* const* d_in, const int* in_sizes, int n_in,
                              void* d_out, int out_size, void* d_ws, size_t ws_size,
                              hipStream_t stream) {
    const float* features = (const float*)d_in[1];
    const float* W_e1     = (const float*)d_in[2];
    const float* b_e1     = (const float*)d_in[3];
    const float* W_e2     = (const float*)d_in[4];
    const float* b_e2     = (const float*)d_in[5];
    const float* W_l0     = (const float*)d_in[6];
    const float* b_l0     = (const float*)d_in[7];
    const float* W_l1     = (const float*)d_in[8];
    const float* b_l1     = (const float*)d_in[9];
    const float* W_d      = (const float*)d_in[10];
    const float* b_d      = (const float*)d_in[11];

    char* ws = (char*)d_ws;
    unsigned short* We1s = (unsigned short*)(ws + 0);
    unsigned short* We2s = (unsigned short*)(ws + 98304);
    unsigned short* Wl0s = (unsigned short*)(ws + 229376);
    unsigned short* Wl1s = (unsigned short*)(ws + 360448);
    float*          sBuf = (float*)(ws + 491520);

    // 1) pre-swizzle weights into bf16 WMMA B-fragment layout
    swizzleB<<<(49152 + 255) / 256, 256, 0, stream>>>(W_e1, We1s, 164, 192, 256);
    swizzleB<<<256, 256, 0, stream>>>(W_e2, We2s, 256, 256, 256);
    swizzleB<<<256, 256, 0, stream>>>(W_l0, Wl0s, 256, 256, 256);
    swizzleB<<<256, 256, 0, stream>>>(W_l1, Wl1s, 256, 256, 256);

    // 2) encoder MLP + fused segment sum : 4 segments per block (65536/4 blocks)
    encoder_kernel<<<65536 / SEGB, 256, 0, stream>>>(
        features, We1s, We2s, b_e1, b_e2, sBuf);

    // 3) residual head + decode : 32 segments per block
    head_kernel<<<65536 / (TS * 16), 256, 0, stream>>>(
        sBuf, Wl0s, Wl1s, b_l0, b_l1, W_d, b_d, (float*)d_out);
}
